// E74BlockDiagCell_23751169147417
// MI455X (gfx1250) — compile-verified, matrix operand order checked by reading, not run on hardware
//
#include <hip/hip_runtime.h>
#include <hip/hip_bf16.h>
#include <math.h>

#define Tn      2048
#define Bn      8
#define Dn      1024
#define NSTATE  1024
#define NBb     128
#define Mrows   (Tn * Bn)   // 16384
#define CH      8           // scan chunk (timesteps staged in LDS)

typedef __bf16 bf16;
typedef __attribute__((ext_vector_type(16))) __bf16 v16bf;
typedef __attribute__((ext_vector_type(8)))  float  v8f;
typedef __attribute__((ext_vector_type(4)))  unsigned int v4u;
typedef __attribute__((ext_vector_type(8)))  int    v8i;
typedef __attribute__((ext_vector_type(4)))  int    v4i;

#if __has_builtin(__builtin_amdgcn_tanhf)
#define TANHF(x) __builtin_amdgcn_tanhf(x)
#else
#define TANHF(x) tanhf(x)
#endif

#if defined(__HIP_DEVICE_COMPILE__) && __has_builtin(__builtin_amdgcn_tensor_load_to_lds) && \
    __has_builtin(__builtin_amdgcn_s_wait_tensorcnt)
#define USE_TDM 1
#else
#define USE_TDM 0
#endif

// ---------------------------------------------------------------- fp32->bf16
__global__ __launch_bounds__(256)
void cvt_bf16_kernel(const float* __restrict__ src, bf16* __restrict__ dst, int n) {
    int i = blockIdx.x * blockDim.x + threadIdx.x;
    int stride = gridDim.x * blockDim.x;
    for (; i < n; i += stride) dst[i] = (bf16)src[i];
}

// ---------------------------------------------------------------- GEMM: Y = X (M x K) * W^T (N x K)
// 32x32 output per wave: 2 A-frags x 2 B-frags -> 4 accumulators.
// Both operands K-major: identical per-lane fragment layout (ISA 7.12.2):
// lane L holds row (L&15); K elements [8*(L>>4)..+7] and [16+8*(L>>4)..+7].
__global__ __launch_bounds__(256)
void gemm_bf16_kernel(const bf16* __restrict__ X, const bf16* __restrict__ W,
                      float* __restrict__ Y) {
    const int lane = threadIdx.x & 31;
    const int wave = threadIdx.x >> 5;
    const int tile = blockIdx.x * 8 + wave;      // 0..16383
    const int tm = tile >> 5;                    // M/32 index (0..511)
    const int tn = tile & 31;                    // N/32 index (0..31)
    const int r16 = lane & 15;
    const int hi  = lane >> 4;

    const bf16* xa0 = X + (size_t)(tm * 32 + r16) * Dn + hi * 8;
    const bf16* xa1 = xa0 + (size_t)16 * Dn;
    const bf16* wa0 = W + (size_t)(tn * 32 + r16) * Dn + hi * 8;
    const bf16* wa1 = wa0 + (size_t)16 * Dn;

    union Frag { v16bf v; uint4 q[2]; };
    v8f c00 = {}, c01 = {}, c10 = {}, c11 = {};

#pragma unroll 2
    for (int k0 = 0; k0 < Dn; k0 += 32) {
        Frag a0, a1, b0, b1;
        a0.q[0] = *(const uint4*)(xa0 + k0);
        a0.q[1] = *(const uint4*)(xa0 + k0 + 16);
        a1.q[0] = *(const uint4*)(xa1 + k0);
        a1.q[1] = *(const uint4*)(xa1 + k0 + 16);
        b0.q[0] = *(const uint4*)(wa0 + k0);
        b0.q[1] = *(const uint4*)(wa0 + k0 + 16);
        b1.q[0] = *(const uint4*)(wa1 + k0);
        b1.q[1] = *(const uint4*)(wa1 + k0 + 16);
#if defined(__HIP_DEVICE_COMPILE__)
        c00 = __builtin_amdgcn_wmma_f32_16x16x32_bf16(false, a0.v, false, b0.v, (short)0, c00, false, false);
        c01 = __builtin_amdgcn_wmma_f32_16x16x32_bf16(false, a0.v, false, b1.v, (short)0, c01, false, false);
        c10 = __builtin_amdgcn_wmma_f32_16x16x32_bf16(false, a1.v, false, b0.v, (short)0, c10, false, false);
        c11 = __builtin_amdgcn_wmma_f32_16x16x32_bf16(false, a1.v, false, b1.v, (short)0, c11, false, false);
#endif
    }

    // C/D layout: VGPR r -> M = r + 8*hi, N = r16
    const int row0 = tm * 32 + hi * 8;
    const int col0 = tn * 32 + r16;
    float* y00 = Y + (size_t)row0 * NSTATE + col0;
    float* y01 = y00 + 16;
    float* y10 = Y + (size_t)(row0 + 16) * NSTATE + col0;
    float* y11 = y10 + 16;
#pragma unroll
    for (int r = 0; r < 8; ++r) {
        y00[(size_t)r * NSTATE] = c00[r];
        y01[(size_t)r * NSTATE] = c01[r];
        y10[(size_t)r * NSTATE] = c10[r];
        y11[(size_t)r * NSTATE] = c11[r];
    }
}

// ---------------------------------------------------------------- TDM 2-D tile load
#if USE_TDM
// Copy an 8-row x 256-float tile (row stride 8192 floats) from gsrc into LDS
// at byte offset lds_off. D# per ISA cdna5 08_async_tensor §8.3/8.4.
__device__ __forceinline__ void tdm_load_2d(const float* gsrc, unsigned int lds_off) {
    unsigned long long ga = (unsigned long long)(uintptr_t)gsrc;
    v4u g0;
    g0[0] = 1u;                                   // count=1, user descriptor
    g0[1] = lds_off;                              // lds_addr (bytes)
    g0[2] = (unsigned int)ga;                     // global_addr[31:0]
    g0[3] = (unsigned int)((ga >> 32) & 0x1FFFFFFu) | (2u << 30);  // addr[56:32] | type=2
    v8i g1;
    g1[0] = (int)(2u << 16);                      // wg_mask=0, data_size=2 (4B), no flags
    g1[1] = (int)(256u << 16);                    // tensor_dim0 = 256 (lo16 in [31:16])
    g1[2] = (int)((unsigned)CH << 16);            // tensor_dim0 hi16=0 | tensor_dim1=CH lo16
    g1[3] = (int)(256u << 16);                    // tensor_dim1 hi16=0 | tile_dim0=256
    g1[4] = (int)CH;                              // tile_dim1=CH | tile_dim2=0
    g1[5] = (int)(Bn * NSTATE);                   // tensor_dim0_stride = 8192 (lo32)
    g1[6] = 0;                                    // stride0 hi16 | stride1 lo16
    g1[7] = 0;                                    // stride1 hi32
    v4i z = {0, 0, 0, 0};
#if __clang_major__ >= 23
    v8i z8 = {0, 0, 0, 0, 0, 0, 0, 0};
    __builtin_amdgcn_tensor_load_to_lds(g0, g1, z, z, z8, 0);
#else
    __builtin_amdgcn_tensor_load_to_lds(g0, g1, z, z, 0);
#endif
}
#endif

// ---------------------------------------------------------------- recurrent scan
// 32 blocks x 256 threads. Block -> (b = bid>>2, nb0 = (bid&3)*32): 32 chains.
// Thread: chain c = tid>>3, row i = tid&7. Lane i holds S row i (8 VGPRs).
// k/v/q staged through double-buffered LDS; TDM-driven when available.
__global__ __launch_bounds__(256)
void scan_kernel(const float* __restrict__ kf, const float* __restrict__ vf,
                 const float* __restrict__ qf, float* __restrict__ out) {
    __shared__ float lds[2][3][CH * 256];

    const int tid = threadIdx.x;
    const int b   = blockIdx.x >> 2;
    const int nb0 = (blockIdx.x & 3) * 32;
    const int c   = tid >> 3;
    const int i   = tid & 7;
    const int nb  = nb0 + c;

    float S[8];
#pragma unroll
    for (int j = 0; j < 8; ++j) S[j] = 0.f;

    const float* srcs[3] = { kf, vf, qf };
    const size_t chainOff = (size_t)b * NSTATE + nb0 * 8;   // per-t0 add t0*Bn*NSTATE

#if USE_TDM
    // wave 0 drives the TDM; everyone else just computes.
    const bool tdmWave = (tid >> 5) == 0;
    if (tdmWave) {
#pragma unroll
        for (int arr = 0; arr < 3; ++arr)
            tdm_load_2d(srcs[arr] + chainOff,
                        (unsigned int)(uintptr_t)&lds[0][arr][0]);
        __builtin_amdgcn_s_wait_tensorcnt(0);
    }
    __syncthreads();
#else
    float4 pre[6];
    auto loadRegs = [&](int t0) {
#pragma unroll
        for (int it = 0; it < 6; ++it) {
            int u   = tid + it * 256;          // 0..1535
            int arr = u >> 9;                  // / (CH*64)
            int rem = u & 511;
            int st  = rem >> 6;
            int seg = rem & 63;
            pre[it] = *(const float4*)(srcs[arr] + (size_t)(t0 + st) * Bn * NSTATE
                                       + chainOff + (seg << 2));
        }
    };
    auto regsToLds = [&](int buf) {
#pragma unroll
        for (int it = 0; it < 6; ++it) {
            int u   = tid + it * 256;
            int arr = u >> 9;
            int rem = u & 511;
            int st  = rem >> 6;
            int seg = rem & 63;
            *(float4*)&lds[buf][arr][st * 256 + (seg << 2)] = pre[it];
        }
    };
    loadRegs(0);
    regsToLds(0);
    __syncthreads();
#endif

    int buf = 0;
    for (int t0 = 0; t0 < Tn; t0 += CH) {
        const bool more = (t0 + CH) < Tn;

#if USE_TDM
        if (more && tdmWave) {
#pragma unroll
            for (int arr = 0; arr < 3; ++arr)
                tdm_load_2d(srcs[arr] + (size_t)(t0 + CH) * Bn * NSTATE + chainOff,
                            (unsigned int)(uintptr_t)&lds[buf ^ 1][arr][0]);
        }
#else
        if (more) loadRegs(t0 + CH);           // prefetch next chunk into VGPRs
#endif

#pragma unroll
        for (int st = 0; st < CH; ++st) {
            const float* kk = &lds[buf][0][st * 256 + c * 8];
            const float* qq = &lds[buf][2][st * 256 + c * 8];
            float vi = lds[buf][1][st * 256 + c * 8 + i];
            float4 kA = *(const float4*)kk;
            float4 kB = *(const float4*)(kk + 4);
            float4 qA = *(const float4*)qq;
            float4 qB = *(const float4*)(qq + 4);
            float kr[8] = { kA.x, kA.y, kA.z, kA.w, kB.x, kB.y, kB.z, kB.w };
            float qr[8] = { qA.x, qA.y, qA.z, qA.w, qB.x, qB.y, qB.z, qB.w };

            float ss = 0.f;
#pragma unroll
            for (int j = 0; j < 8; ++j) ss += kr[j] * kr[j];
            float inv = 1.f / (sqrtf(ss) + 1e-6f);

            float kn[8];
            float retr = 0.f;
#pragma unroll
            for (int j = 0; j < 8; ++j) { kn[j] = kr[j] * inv; retr += S[j] * kn[j]; }
            float delta = vi - retr;

            float sq = 0.f;
#pragma unroll
            for (int j = 0; j < 8; ++j) {
                S[j] = TANHF(S[j] + delta * kn[j]);
                sq += S[j] * qr[j];
            }
            float sig = 1.f / (1.f + __expf(-sq));
            out[((size_t)(t0 + st) * Bn + b) * NSTATE + nb * 8 + i] = sq * sq * sig;
        }

#if USE_TDM
        if (more && tdmWave) __builtin_amdgcn_s_wait_tensorcnt(0);
#else
        if (more) regsToLds(buf ^ 1);          // drain prefetch into other buffer
#endif
        __syncthreads();
        buf ^= 1;
    }

    // final state: S_final[b][nb][i][j]
    float* sf = out + (size_t)Tn * Bn * NSTATE;
    size_t base = (((size_t)b * NBb + nb) * 8 + i) * 8;
#pragma unroll
    for (int j = 0; j < 8; ++j) sf[base + j] = S[j];
}

// ---------------------------------------------------------------- launch
extern "C" void kernel_launch(void* const* d_in, const int* in_sizes, int n_in,
                              void* d_out, int out_size, void* d_ws, size_t ws_size,
                              hipStream_t stream) {
    (void)in_sizes; (void)n_in; (void)out_size; (void)ws_size;
    const float* x  = (const float*)d_in[0];
    const float* Wk = (const float*)d_in[1];
    const float* Wv = (const float*)d_in[2];
    const float* Wq = (const float*)d_in[3];
    float* out = (float*)d_out;

    const size_t Xe = (size_t)Mrows * Dn;     // 16,777,216
    const size_t We = (size_t)NSTATE * Dn;    // 1,048,576

    bf16*  xb  = (bf16*)d_ws;
    bf16*  wkb = xb + Xe;
    bf16*  wvb = wkb + We;
    bf16*  wqb = wvb + We;
    float* kf  = (float*)(wqb + We);          // byte offset 16B-aligned
    float* vf  = kf + Xe;
    float* qf  = vf + Xe;

    cvt_bf16_kernel<<<4096, 256, 0, stream>>>(x,  xb,  (int)Xe);
    cvt_bf16_kernel<<<1024, 256, 0, stream>>>(Wk, wkb, (int)We);
    cvt_bf16_kernel<<<1024, 256, 0, stream>>>(Wv, wvb, (int)We);
    cvt_bf16_kernel<<<1024, 256, 0, stream>>>(Wq, wqb, (int)We);

    // (M/32)x(N/32) = 512*32 = 16384 tiles, 8 waves per block -> 2048 blocks
    gemm_bf16_kernel<<<2048, 256, 0, stream>>>(xb, wkb, kf);
    gemm_bf16_kernel<<<2048, 256, 0, stream>>>(xb, wvb, vf);
    gemm_bf16_kernel<<<2048, 256, 0, stream>>>(xb, wqb, qf);

    scan_kernel<<<32, 256, 0, stream>>>(kf, vf, qf, out);
}
